// LSTMPositionNet_3891240370983
// MI455X (gfx1250) — matrix operand order repaired and checked by reading, not compiled, not run on hardware
//
#include <hip/hip_runtime.h>
#include <hip/hip_bf16.h>

typedef __attribute__((ext_vector_type(16))) _Float16 v16h;
typedef __attribute__((ext_vector_type(8)))  float    v8f;

#define LSTM_F 64
#define LSTM_H 32
#define LSTM_G 128   // 4*H

// ---------------- fast transcendentals ----------------
__device__ inline float ftanh(float x) {
#if defined(__has_builtin)
#if __has_builtin(__builtin_amdgcn_tanhf)
  return __builtin_amdgcn_tanhf(x);   // gfx1250 v_tanh_f32
#else
  return tanhf(x);
#endif
#else
  return tanhf(x);
#endif
}

__device__ inline float fsigm(float x) { return 0.5f + 0.5f * ftanh(0.5f * x); }

__device__ inline v8f vtanh(v8f x) {
  v8f r;
#pragma unroll
  for (int i = 0; i < 8; ++i) r[i] = ftanh(x[i]);
  return r;
}
__device__ inline v8f vsigm(v8f x) {
  v8f r;
#pragma unroll
  for (int i = 0; i < 8; ++i) r[i] = fsigm(x[i]);
  return r;
}

// ---------------- fragment loaders (f32 memory -> f16 fragments) ----------------
// 8 contiguous floats at p -> frag[hbase .. hbase+7]
__device__ inline void load8_cvt(const float* __restrict__ p, v16h& frag, int hbase) {
  float4 a = *reinterpret_cast<const float4*>(p);
  float4 b = *reinterpret_cast<const float4*>(p + 4);
  frag[hbase + 0] = (_Float16)a.x; frag[hbase + 1] = (_Float16)a.y;
  frag[hbase + 2] = (_Float16)a.z; frag[hbase + 3] = (_Float16)a.w;
  frag[hbase + 4] = (_Float16)b.x; frag[hbase + 5] = (_Float16)b.y;
  frag[hbase + 6] = (_Float16)b.z; frag[hbase + 7] = (_Float16)b.w;
}

// A fragment (16x32 f16, M=row via lane%16): two 8-wide K runs per lane.
__device__ inline v16h load_afrag(const float* __restrict__ row, int klo, int khi) {
  v16h f;
  load8_cvt(row + klo, f, 0);
  load8_cvt(row + khi, f, 8);
  return f;
}

// B fragment (32x16 f16, N=col via lane%16): one contiguous 16-wide K run per lane.
__device__ inline v16h load_bfrag(const float* __restrict__ p) {
  v16h f;
  load8_cvt(p, f, 0);
  load8_cvt(p + 8, f, 8);
  return f;
}

// ---------------- fused LSTM + head ----------------
// One wave handles a 16-row batch tile; transposed gate GEMM:
//   G^T[128,16] = W_ih[128,64] * x_t^T[64,16] + W_hh[128,32] * h^T[32,16] + bias
// A-side = weights (resident in VGPRs for all 256 timesteps), B-side = activations.
__global__ __launch_bounds__(256, 1)
void lstm_pos_fused(const float* __restrict__ x,        // [B,T,64]
                    const int*   __restrict__ ticker,   // [B]
                    const float* __restrict__ Wih,      // [128,64]
                    const float* __restrict__ Whh,      // [128,32]
                    const float* __restrict__ bih,      // [128]
                    const float* __restrict__ bhh,      // [128]
                    const float* __restrict__ emb,      // [500,8]
                    const float* __restrict__ fcw,      // [40]
                    const float* __restrict__ fcb,      // [1]
                    float* __restrict__ out,            // [B]
                    int Bsz, int Tlen) {
  const int lane = threadIdx.x & 31;
  const int wave = threadIdx.x >> 5;
  const int gw   = blockIdx.x * (blockDim.x >> 5) + wave;
  const int b0   = gw * 16;
  if (b0 >= Bsz) return;                 // wave-uniform; EXEC stays all-ones

  const int m   = lane & 15;             // A: gate row in tile; B: batch col in tile
  const int hiL = lane >> 4;             // lane half
  const int aklo = hiL ? 16 : 0;         // A-frag K runs per lane half
  const int akhi = hiL ? 24 : 8;
  const int bkb  = hiL ? 16 : 0;         // B-frag K base per lane half

  // ---- resident weight fragments + bias ----
  v16h Aih[8][2];   // 8 gate tiles x 2 K-chunks of W_ih
  v16h Ahh[8];      // 8 gate tiles of W_hh
  v8f  bias[8];
#pragma unroll
  for (int j = 0; j < 8; ++j) {
    const int g = 16 * j + m;
    const float* ri = Wih + g * LSTM_F;
    Aih[j][0] = load_afrag(ri,      aklo, akhi);
    Aih[j][1] = load_afrag(ri + 32, aklo, akhi);
    Ahh[j]    = load_afrag(Whh + g * LSTM_H, aklo, akhi);
#pragma unroll
    for (int v = 0; v < 8; ++v) {
      const int gg = 16 * j + v + 8 * hiL;   // gate row held by (vgpr v, this lane)
      bias[j][v] = bih[gg] + bhh[gg];
    }
  }

  // fc weights matching the h^T register distribution
  v8f fw0, fw1;
#pragma unroll
  for (int v = 0; v < 8; ++v) {
    fw0[v] = fcw[v + 8 * hiL];
    fw1[v] = fcw[16 + v + 8 * hiL];
  }

  // ---- state ----
  v8f c0 = {}, c1 = {}, h0 = {}, h1 = {};    // c^T, h^T tiles (f32, D-layout)
  v16h Bh = {};                              // h^T as B fragment (h=0 at t=0)

  const float* xp = x + (size_t)(b0 + m) * (size_t)Tlen * LSTM_F + bkb;
  v16h Bx0 = load_bfrag(xp);        // t=0, K 0..31 (this lane-half's 16)
  v16h Bx1 = load_bfrag(xp + 32);   // t=0, K 32..63

  for (int t = 0; t < Tlen; ++t) {
    // ---- gate GEMM: 3 WMMAs per gate tile, f32 accumulate seeded with bias ----
    v8f acc[8];
#pragma unroll
    for (int j = 0; j < 8; ++j) {
      v8f a = bias[j];
      a = __builtin_amdgcn_wmma_f32_16x16x32_f16(false, Aih[j][0], false, Bx0,
                                                 (short)0, a, false, false);
      a = __builtin_amdgcn_wmma_f32_16x16x32_f16(false, Aih[j][1], false, Bx1,
                                                 (short)0, a, false, false);
      a = __builtin_amdgcn_wmma_f32_16x16x32_f16(false, Ahh[j],    false, Bh,
                                                 (short)0, a, false, false);
      acc[j] = a;
    }

    // ---- software pipeline: issue next x-tile loads before the VALU tail ----
    if (t + 1 < Tlen) {
      const float* nx = xp + (size_t)(t + 1) * LSTM_F;
      Bx0 = load_bfrag(nx);
      Bx1 = load_bfrag(nx + 32);
      if (t + 2 < Tlen)
        __builtin_prefetch(xp + (size_t)(t + 2) * LSTM_F, 0, 1);  // global_prefetch_b8
    }

    // ---- i,f,g,o (tiles 0..7) -> c,h update, all layout-aligned elementwise ----
    v8f i0 = vsigm(acc[0]), i1 = vsigm(acc[1]);
    v8f f0 = vsigm(acc[2]), f1 = vsigm(acc[3]);
    v8f g0 = vtanh(acc[4]), g1 = vtanh(acc[5]);
    v8f o0 = vsigm(acc[6]), o1 = vsigm(acc[7]);
    c0 = f0 * c0 + i0 * g0;
    c1 = f1 * c1 + i1 * g1;
    h0 = o0 * vtanh(c0);
    h1 = o1 * vtanh(c1);

    // ---- rebuild h^T B-fragment: lane-half swap (no LDS) ----
#pragma unroll
    for (int v = 0; v < 8; ++v) {
      float s0 = __shfl_xor(h0[v], 16, 32);
      float s1 = __shfl_xor(h1[v], 16, 32);
      float lo = hiL ? s1 : h0[v];     // K = bkb + v
      float hv = hiL ? h1[v] : s0;     // K = bkb + 8 + v
      Bh[v]     = (_Float16)lo;
      Bh[8 + v] = (_Float16)hv;
    }
  }

  // ---- head: pos = tanh(h . fcw[0:32] + emb[ticker] . fcw[32:40] + fcb) ----
  float part = 0.f;
#pragma unroll
  for (int v = 0; v < 8; ++v) part += h0[v] * fw0[v] + h1[v] * fw1[v];
  part += __shfl_xor(part, 16, 32);     // combine the two lane-half K ranges
  if (hiL == 0) {
    const int b   = b0 + m;
    const int tid = ticker[b];
    float s = part + fcb[0];
#pragma unroll
    for (int e = 0; e < 8; ++e) s += emb[tid * 8 + e] * fcw[32 + e];
    out[b] = ftanh(s);
  }
}

extern "C" void kernel_launch(void* const* d_in, const int* in_sizes, int n_in,
                              void* d_out, int out_size, void* d_ws, size_t ws_size,
                              hipStream_t stream) {
  const float* x      = (const float*)d_in[0];
  const int*   ticker = (const int*)d_in[1];
  const float* Wih    = (const float*)d_in[2];
  const float* Whh    = (const float*)d_in[3];
  const float* bih    = (const float*)d_in[4];
  const float* bhh    = (const float*)d_in[5];
  const float* emb    = (const float*)d_in[6];
  const float* fcw    = (const float*)d_in[7];
  const float* fcb    = (const float*)d_in[8];
  float* out = (float*)d_out;

  const int Bsz  = in_sizes[1];                       // 4096
  const int Tlen = in_sizes[0] / (Bsz * LSTM_F);      // 256

  const int waves  = (Bsz + 15) / 16;                 // 16 batch rows per wave
  const int wpb    = 8;                               // 256 threads
  const int blocks = (waves + wpb - 1) / wpb;

  lstm_pos_fused<<<dim3(blocks), dim3(wpb * 32), 0, stream>>>(
      x, ticker, Wih, Whh, bih, bhh, emb, fcw, fcb, out, Bsz, Tlen);
}
